// InformationalFlowLayer_28939489640901
// MI455X (gfx1250) — matrix operand order repaired and checked by reading, not compiled, not run on hardware
//
#include <hip/hip_runtime.h>
#include <math.h>

typedef __bf16 bf16;
typedef __attribute__((ext_vector_type(16))) __bf16    v16bf;
typedef __attribute__((ext_vector_type(8)))  float     v8f;
typedef __attribute__((ext_vector_type(4)))  unsigned  u32x4;
typedef __attribute__((ext_vector_type(8)))  int       i32x8;
typedef unsigned int u32;

#define HIDDEN   1024
#define HEADS    16
#define HEAD_DIM 64
#define BATCH    2
#define SEQ      2048
#define ROWS     (BATCH*SEQ)          // 4096
#define LN_EPS   1e-5f

__device__ __forceinline__ v8f vzero8() {
    v8f z = {0.f,0.f,0.f,0.f,0.f,0.f,0.f,0.f};
    return z;
}
__device__ __forceinline__ float hmax16(float v) {
    #pragma unroll
    for (int o = 1; o < 16; o <<= 1) v = fmaxf(v, __shfl_xor(v, o, 32));
    return v;
}
__device__ __forceinline__ float hsum16(float v) {
    #pragma unroll
    for (int o = 1; o < 16; o <<= 1) v += __shfl_xor(v, o, 32);
    return v;
}
__device__ __forceinline__ v8f wmma_bf16(v16bf a, v16bf b, v8f c) {
    return __builtin_amdgcn_wmma_f32_16x16x32_bf16(false, a, false, b,
                                                   (short)0, c, false, false);
}
// A fragment (16x32 bf16): lane = row, 16 contiguous K per half
__device__ __forceinline__ v16bf load_a_frag(const bf16* rowp, int half) {
    v16bf a;
    const bf16* p = rowp + half * 16;
    #pragma unroll
    for (int j = 0; j < 16; ++j) a[j] = p[j];
    return a;
}
// B fragment (32x16 bf16): lane = column, 16 contiguous K per half
__device__ __forceinline__ v16bf load_b_frag(const bf16* colp) {
    v16bf b;
    #pragma unroll
    for (int j = 0; j < 16; ++j) b[j] = colp[j];
    return b;
}

// ---------------------------------------------------------------------------
// Tensor Data Mover: async 2D bf16 tile copy global -> LDS (ISA ch.8 D#).
// Issued by one wave (scalar-branch guarded); EXEC ignored; TENSORcnt-tracked.
// ---------------------------------------------------------------------------
__device__ __forceinline__ void tdm_load_2d_bf16(u32 lds_off, const bf16* gptr,
                                                 u32 tile_k, u32 tile_rows,
                                                 u32 stride_elems)
{
    const unsigned long long ga = (unsigned long long)(uintptr_t)gptr;
    u32x4 g0;
    g0[0] = 1u;                                      // count=1, user descriptor
    g0[1] = lds_off;                                 // LDS byte address
    g0[2] = (u32)ga;                                 // global_addr[31:0]
    g0[3] = (u32)((ga >> 32) & 0x01FFFFFFu) | (2u << 30);  // addr[56:32] | type=2
    const u32 dim0 = stride_elems;                   // tensor width (elems)
    const u32 dim1 = tile_rows;
    i32x8 g1;
    g1[0] = (int)(1u << 16);                                      // data_size=2B
    g1[1] = (int)((dim0 & 0xFFFFu) << 16);                        // dim0[15:0]
    g1[2] = (int)(((dim0 >> 16) & 0xFFFFu) | ((dim1 & 0xFFFFu) << 16));
    g1[3] = (int)(((dim1 >> 16) & 0xFFFFu) | ((tile_k & 0xFFFFu) << 16)); // tile_dim0
    g1[4] = (int)(tile_rows & 0xFFFFu);                           // tile_dim1
    g1[5] = (int)stride_elems;                                    // dim0_stride[31:0]
    g1[6] = 0;
    g1[7] = 0;
    asm volatile("tensor_load_to_lds %0, %1" :: "s"(g0), "s"(g1) : "memory");
}
__device__ __forceinline__ u32 lds_off32(const void* p) {
    return (u32)(uintptr_t)p;    // low 32 bits of generic addr = LDS offset
}

// ---------------------------------------------------------------------------
// Prep kernels: one-time bf16 conversion (amortizes cvt across all tile reuse)
// ---------------------------------------------------------------------------
__global__ __launch_bounds__(256) void convert_bf16(const float* __restrict__ src,
                                                    bf16* __restrict__ dst)
{
    const int idx = (blockIdx.x * 256 + threadIdx.x) * 8;
    __builtin_prefetch(src + idx + 256 * 8, 0, 1);
    #pragma unroll
    for (int j = 0; j < 8; ++j) dst[idx + j] = (bf16)src[idx + j];
}

// W [K][N] f32 -> WT [N][K] bf16 (tiled transpose through LDS)
__global__ __launch_bounds__(256) void transpose_to_bf16(const float* __restrict__ W,
                                                         bf16* __restrict__ WT,
                                                         int Kd, int Nd)
{
    __shared__ bf16 tile[64][65];
    const int t = threadIdx.x;
    const int r = t >> 2, c0 = (t & 3) * 16;
    const int n0 = blockIdx.x * 64, k0 = blockIdx.y * 64;
    const float* src = W + (size_t)(k0 + r) * Nd + n0 + c0;
    #pragma unroll
    for (int j = 0; j < 16; ++j) tile[c0 + j][r] = (bf16)src[j];
    __syncthreads();
    bf16* dst = WT + (size_t)(n0 + r) * Kd + k0 + c0;
    #pragma unroll
    for (int j = 0; j < 16; ++j) dst[j] = tile[r][c0 + j];
}

// ---------------------------------------------------------------------------
// Shared GEMM body: C[128x64] tile, k-step 64, double-buffered TDM staging.
// Each of 4 waves computes 32x64 (16 WMMAs per k-step).
// ---------------------------------------------------------------------------
__device__ __forceinline__ void gemm_tile_128x64(
    const bf16* __restrict__ Asrc0, const bf16* __restrict__ Asrc1, int ksplit,
    const bf16* __restrict__ Bsrc, u32 bstride, int Ktot,
    int wave, int row, int half, v8f acc[2][4],
    bf16* Xs /* [2][128*64] */, bf16* Wt /* [2][64*64] */)
{
    const u32 xOff0 = lds_off32(Xs), xOff1 = xOff0 + 128 * 64 * 2;
    const u32 wOff0 = lds_off32(Wt), wOff1 = wOff0 + 64 * 64 * 2;

    if (wave == 0) {   // prologue: fill buffer 0
        tdm_load_2d_bf16(xOff0, Asrc0, 64, 128, HIDDEN);
        tdm_load_2d_bf16(wOff0, Bsrc, 64, 64, bstride);
    }
    for (int kk = 0; kk < Ktot; kk += 64) {
        const int cur = (kk >> 6) & 1;
        if (wave == 0) {
            if (kk + 64 < Ktot) {   // kick next tile into the other buffer
                const int kn = kk + 64;
                const bf16* an = (kn < ksplit) ? Asrc0 + kn : Asrc1 + (kn - ksplit);
                tdm_load_2d_bf16(cur ? xOff0 : xOff1, an, 64, 128, HIDDEN);
                tdm_load_2d_bf16(cur ? wOff0 : wOff1, Bsrc + kn, 64, 64, bstride);
                __builtin_amdgcn_s_wait_tensorcnt(2);  // current pair has landed
            } else {
                __builtin_amdgcn_s_wait_tensorcnt(0);
            }
        }
        __syncthreads();   // publish current buffer

        const bf16* Xc = Xs + cur * (128 * 64);
        const bf16* Wc = Wt + cur * (64 * 64);
        #pragma unroll
        for (int c = 0; c < 2; ++c) {
            v16bf A[2], B[4];
            #pragma unroll
            for (int rt = 0; rt < 2; ++rt)
                A[rt] = load_a_frag(&Xc[(wave * 32 + rt * 16 + row) * 64 + c * 32], half);
            #pragma unroll
            for (int ct = 0; ct < 4; ++ct)
                B[ct] = load_b_frag(&Wc[(ct * 16 + row) * 64 + c * 32 + half * 16]);
            #pragma unroll
            for (int rt = 0; rt < 2; ++rt)
                #pragma unroll
                for (int ct = 0; ct < 4; ++ct)
                    acc[rt][ct] = wmma_bf16(A[rt], B[ct], acc[rt][ct]);
        }
        __syncthreads();   // retire current buffer before it is re-filled
    }
}

// ---------------------------------------------------------------------------
// Kernel 1: QKV projections. grid (N/64=16, M/128=32, 3), block 128 (4 waves).
// ---------------------------------------------------------------------------
__global__ __launch_bounds__(128) void qkv_gemm(
    const bf16* __restrict__ xb,
    const bf16* __restrict__ WqT, const bf16* __restrict__ WkT, const bf16* __restrict__ WvT,
    const float* __restrict__ bq, const float* __restrict__ bk, const float* __restrict__ bv,
    bf16* __restrict__ Q, bf16* __restrict__ K, bf16* __restrict__ V)
{
    __shared__ bf16 Xs[2][128 * 64];   // 32 KB double-buffered A tiles
    __shared__ bf16 Wt[2][64 * 64];    // 16 KB double-buffered B tiles

    const int tid  = threadIdx.x;
    const int wave = __builtin_amdgcn_readfirstlane(tid >> 5);  // SGPR-uniform
    const int lane = tid & 31;
    const int row  = lane & 15, half = lane >> 4;
    const int z    = blockIdx.z;
    const bf16*  WT   = (z == 0) ? WqT : (z == 1) ? WkT : WvT;
    const float* bias = (z == 0) ? bq  : (z == 1) ? bk  : bv;
    bf16*        Out  = (z == 0) ? Q   : (z == 1) ? K   : V;
    const int n0 = blockIdx.x * 64, m0 = blockIdx.y * 128;

    v8f acc[2][4];
    #pragma unroll
    for (int rt = 0; rt < 2; ++rt)
        #pragma unroll
        for (int ct = 0; ct < 4; ++ct) acc[rt][ct] = vzero8();

    const bf16* Asrc = xb + (size_t)m0 * HIDDEN;
    const bf16* Bsrc = WT + (size_t)n0 * HIDDEN;
    gemm_tile_128x64(Asrc, Asrc, HIDDEN, Bsrc, HIDDEN, HIDDEN,
                     wave, row, half, acc, &Xs[0][0], &Wt[0][0]);

    #pragma unroll
    for (int ct = 0; ct < 4; ++ct) {
        const int n = n0 + ct * 16 + row;
        const float bv_ = bias[n];
        #pragma unroll
        for (int rt = 0; rt < 2; ++rt)
            #pragma unroll
            for (int r = 0; r < 8; ++r) {
                const int m = m0 + wave * 32 + rt * 16 + half * 8 + r;
                Out[(size_t)m * HIDDEN + n] = (bf16)(acc[rt][ct][r] + bv_);
            }
    }
}

// ---------------------------------------------------------------------------
// Kernel 2: flash-style antisymmetric attention: flow = Q·K^T - K·Q^T.
// grid (S/64=32, HEADS, B), block 128. K/Q key-blocks double-buffered via TDM.
// ---------------------------------------------------------------------------
__global__ __launch_bounds__(128) void attn(
    const bf16* __restrict__ Q, const bf16* __restrict__ K,
    const bf16* __restrict__ V, float* __restrict__ flowed,
    bf16* __restrict__ flowedb)
{
    __shared__ bf16 Kls[2][32 * 64];   // 8 KB double-buffered key-block K rows
    __shared__ bf16 Qls[2][32 * 64];   // 8 KB double-buffered key-block Q rows
    __shared__ bf16 wls[4][16][32];    // 4 KB per-wave softmax weights

    const int tid  = threadIdx.x;
    const int wave = __builtin_amdgcn_readfirstlane(tid >> 5);
    const int lane = tid & 31;
    const int row  = lane & 15, half = lane >> 4;
    const int qb = blockIdx.x * 64 + wave * 16;
    const int h  = blockIdx.y, b = blockIdx.z;
    const size_t base = ((size_t)b * SEQ) * HIDDEN + (size_t)h * HEAD_DIM;

    const u32 kOff0 = lds_off32(&Kls[0][0]), kOff1 = kOff0 + 32 * 64 * 2;
    const u32 qOff0 = lds_off32(&Qls[0][0]), qOff1 = qOff0 + 32 * 64 * 2;

    if (wave == 0) {   // prologue: start DMA of the first key block
        tdm_load_2d_bf16(kOff0, K + base, 64, 32, HIDDEN);
        tdm_load_2d_bf16(qOff0, Q + base, 64, 32, HIDDEN);
    }

    v16bf aQ[2], aK[2];
    {
        const int s = qb + row;
        const bf16* qp = Q + base + (size_t)s * HIDDEN;
        const bf16* kp = K + base + (size_t)s * HIDDEN;
        #pragma unroll
        for (int c = 0; c < 2; ++c) {
            aQ[c] = load_a_frag(qp + c * 32, half);
            aK[c] = load_a_frag(kp + c * 32, half);
        }
    }

    float m8[8], l8[8];
    v8f acc[4];
    #pragma unroll
    for (int r = 0; r < 8; ++r) { m8[r] = -3.0e38f; l8[r] = 0.f; }
    #pragma unroll
    for (int j = 0; j < 4; ++j) acc[j] = vzero8();

    const float scale = 0.125f;   // 1/sqrt(64)

    for (int kb = 0; kb < SEQ; kb += 32) {
        const int cur = (kb >> 5) & 1;
        if (wave == 0) {
            if (kb + 32 < SEQ) {
                const bf16* kn = K + base + (size_t)(kb + 32) * HIDDEN;
                const bf16* qn = Q + base + (size_t)(kb + 32) * HIDDEN;
                tdm_load_2d_bf16(cur ? kOff0 : kOff1, kn, 64, 32, HIDDEN);
                tdm_load_2d_bf16(cur ? qOff0 : qOff1, qn, 64, 32, HIDDEN);
                __builtin_amdgcn_s_wait_tensorcnt(2);
            } else {
                __builtin_amdgcn_s_wait_tensorcnt(0);
            }
        }
        __syncthreads();   // current key block ready

        v8f fl[2];
        #pragma unroll
        for (int t = 0; t < 2; ++t) {                 // two 16-key subtiles
            const int kl = t * 16 + row;              // lane = column (key)
            v8f s1 = vzero8(), s2 = vzero8();
            #pragma unroll
            for (int c = 0; c < 2; ++c) {
                const v16bf bK = load_b_frag(&Kls[cur][kl * 64 + c * 32 + half * 16]);
                const v16bf bQ = load_b_frag(&Qls[cur][kl * 64 + c * 32 + half * 16]);
                s1 = wmma_bf16(aQ[c], bK, s1);        // Q·K^T
                s2 = wmma_bf16(aK[c], bQ, s2);        // K·Q^T
            }
            #pragma unroll
            for (int r = 0; r < 8; ++r) fl[t][r] = (s1[r] - s2[r]) * scale;
        }

        #pragma unroll
        for (int r = 0; r < 8; ++r) {                 // online softmax
            const float tmax = hmax16(fmaxf(fl[0][r], fl[1][r]));
            const float mnew = fmaxf(m8[r], tmax);
            const float alpha = __expf(m8[r] - mnew);
            const float e0 = __expf(fl[0][r] - mnew);
            const float e1 = __expf(fl[1][r] - mnew);
            l8[r] = l8[r] * alpha + hsum16(e0 + e1);
            m8[r] = mnew;
            #pragma unroll
            for (int j = 0; j < 4; ++j) acc[j][r] *= alpha;
            const int m = half * 8 + r;
            wls[wave][m][row]      = (bf16)e0;
            wls[wave][m][16 + row] = (bf16)e1;
        }
        __syncthreads();   // wls publish + retire current K/Q buffer reads

        const v16bf aW = load_a_frag(&wls[wave][row][0], half);
        #pragma unroll
        for (int j = 0; j < 4; ++j) {                 // w @ V (V read transposed)
            const int d = j * 16 + row;
            const bf16* vp = V + base + (size_t)(kb + half * 16) * HIDDEN + d;
            v16bf bV;
            #pragma unroll
            for (int i = 0; i < 16; ++i) bV[i] = vp[(size_t)i * HIDDEN];
            acc[j] = wmma_bf16(aW, bV, acc[j]);
        }
    }

    #pragma unroll
    for (int j = 0; j < 4; ++j) {
        const int d = j * 16 + row;
        #pragma unroll
        for (int r = 0; r < 8; ++r) {
            const int q = qb + half * 8 + r;
            const float o = acc[j][r] / l8[r];
            const size_t idx = ((size_t)b * SEQ + q) * HIDDEN + (size_t)h * HEAD_DIM + d;
            flowed[idx]  = o;
            flowedb[idx] = (bf16)o;
        }
    }
}

// ---------------------------------------------------------------------------
// Kernel 3: gate GEMM [4096 x 2048] @ Wg -> gpre (f32). grid (16, 32), block 128
// ---------------------------------------------------------------------------
__global__ __launch_bounds__(128) void gate_gemm(
    const bf16* __restrict__ xb, const bf16* __restrict__ flowedb,
    const bf16* __restrict__ WgT, const float* __restrict__ bg,
    float* __restrict__ gpre)
{
    __shared__ bf16 Xs[2][128 * 64];
    __shared__ bf16 Wt[2][64 * 64];

    const int tid  = threadIdx.x;
    const int wave = __builtin_amdgcn_readfirstlane(tid >> 5);
    const int lane = tid & 31;
    const int row  = lane & 15, half = lane >> 4;
    const int n0 = blockIdx.x * 64, m0 = blockIdx.y * 128;

    v8f acc[2][4];
    #pragma unroll
    for (int rt = 0; rt < 2; ++rt)
        #pragma unroll
        for (int ct = 0; ct < 4; ++ct) acc[rt][ct] = vzero8();

    // A spans the concat: first 1024 k from xb, next 1024 from flowedb
    gemm_tile_128x64(xb + (size_t)m0 * HIDDEN, flowedb + (size_t)m0 * HIDDEN, HIDDEN,
                     WgT + (size_t)n0 * (2 * HIDDEN), 2 * HIDDEN, 2 * HIDDEN,
                     wave, row, half, acc, &Xs[0][0], &Wt[0][0]);

    #pragma unroll
    for (int ct = 0; ct < 4; ++ct) {
        const int n = n0 + ct * 16 + row;
        const float bv_ = bg[n];
        #pragma unroll
        for (int rt = 0; rt < 2; ++rt)
            #pragma unroll
            for (int r = 0; r < 8; ++r) {
                const int m = m0 + wave * 32 + rt * 16 + half * 8 + r;
                gpre[(size_t)m * HIDDEN + n] = acc[rt][ct][r] + bv_;
            }
    }
}

// ---------------------------------------------------------------------------
// Kernel 4: sigmoid gate + residual + LayerNorm.  grid 4096, block 256
// ---------------------------------------------------------------------------
__global__ __launch_bounds__(256) void finalize(
    const float* __restrict__ x, const float* __restrict__ flowed,
    const float* __restrict__ gpre,
    const float* __restrict__ gamma, const float* __restrict__ beta,
    float* __restrict__ out)
{
    __shared__ float sh_s[8], sh_ss[8];
    const int m = blockIdx.x, t = threadIdx.x;
    const size_t rowb = (size_t)m * HIDDEN;

    float yv[4], s = 0.f, ss = 0.f;
    #pragma unroll
    for (int i = 0; i < 4; ++i) {
        const int idx = t * 4 + i;
        const float f = flowed[rowb + idx];
        const float g = 1.f / (1.f + __expf(-gpre[rowb + idx]));
        const float y = x[rowb + idx] + g * f;
        yv[i] = y; s += y; ss += y * y;
    }
    #pragma unroll
    for (int o = 1; o < 32; o <<= 1) { s += __shfl_xor(s, o, 32); ss += __shfl_xor(ss, o, 32); }
    if ((t & 31) == 0) { sh_s[t >> 5] = s; sh_ss[t >> 5] = ss; }
    __syncthreads();
    float ts = 0.f, tss = 0.f;
    #pragma unroll
    for (int w = 0; w < 8; ++w) { ts += sh_s[w]; tss += sh_ss[w]; }
    const float mu   = ts * (1.f / HIDDEN);
    const float var  = tss * (1.f / HIDDEN) - mu * mu;
    const float rstd = rsqrtf(var + LN_EPS);
    #pragma unroll
    for (int i = 0; i < 4; ++i) {
        const int idx = t * 4 + i;
        out[rowb + idx] = (yv[i] - mu) * rstd * gamma[idx] + beta[idx];
    }
}

// ---------------------------------------------------------------------------
extern "C" void kernel_launch(void* const* d_in, const int* in_sizes, int n_in,
                              void* d_out, int out_size, void* d_ws, size_t ws_size,
                              hipStream_t stream)
{
    const float* x  = (const float*)d_in[0];
    const float* Wq = (const float*)d_in[1];
    const float* bq = (const float*)d_in[2];
    const float* Wk = (const float*)d_in[3];
    const float* bk = (const float*)d_in[4];
    const float* Wv = (const float*)d_in[5];
    const float* bv = (const float*)d_in[6];
    const float* Wg = (const float*)d_in[7];
    const float* bg = (const float*)d_in[8];
    const float* gamma = (const float*)d_in[9];
    const float* beta  = (const float*)d_in[10];
    float* out = (float*)d_out;

    const size_t MB = 1024 * 1024;
    const size_t NE = (size_t)ROWS * HIDDEN;                 // 4M elements
    char* ws = (char*)d_ws;
    bf16*  Qw      = (bf16*)(ws);                            //  0 ..  8 MB
    bf16*  Kw      = (bf16*)(ws + 8  * MB);                  //  8 .. 16 MB
    bf16*  Vw      = (bf16*)(ws + 16 * MB);                  // 16 .. 24 MB
    float* flowed  = (float*)(ws + 24 * MB);                 // 24 .. 40 MB
    bf16*  flowedb = (bf16*)(ws + 40 * MB);                  // 40 .. 48 MB
    bf16*  xb      = (bf16*)(ws + 48 * MB);                  // 48 .. 56 MB
    bf16*  WqT     = (bf16*)(ws + 56 * MB);                  // 56 .. 58 MB
    bf16*  WkT     = (bf16*)(ws + 58 * MB);                  // 58 .. 60 MB
    bf16*  WvT     = (bf16*)(ws + 60 * MB);                  // 60 .. 62 MB
    bf16*  WgT     = (bf16*)(ws + 62 * MB);                  // 62 .. 66 MB
    float* gpre    = (float*)(ws);      // aliases Q/K (dead after attention)

    convert_bf16<<<dim3(NE / (256 * 8)), 256, 0, stream>>>(x, xb);
    transpose_to_bf16<<<dim3(16, 16), 256, 0, stream>>>(Wq, WqT, HIDDEN, HIDDEN);
    transpose_to_bf16<<<dim3(16, 16), 256, 0, stream>>>(Wk, WkT, HIDDEN, HIDDEN);
    transpose_to_bf16<<<dim3(16, 16), 256, 0, stream>>>(Wv, WvT, HIDDEN, HIDDEN);
    transpose_to_bf16<<<dim3(16, 32), 256, 0, stream>>>(Wg, WgT, 2 * HIDDEN, HIDDEN);

    qkv_gemm<<<dim3(HIDDEN / 64, ROWS / 128, 3), 128, 0, stream>>>(
        xb, WqT, WkT, WvT, bq, bk, bv, Qw, Kw, Vw);

    attn<<<dim3(SEQ / 64, HEADS, BATCH), 128, 0, stream>>>(Qw, Kw, Vw, flowed, flowedb);

    gate_gemm<<<dim3(HIDDEN / 64, ROWS / 128), 128, 0, stream>>>(
        xb, flowedb, WgT, bg, gpre);

    finalize<<<dim3(ROWS), 256, 0, stream>>>(x, flowed, gpre, gamma, beta, out);
}
